// EdgeBlock_11527692222555
// MI455X (gfx1250) — matrix operand order repaired and checked by reading, not compiled, not run on hardware
//
#include <hip/hip_runtime.h>

// ---------------------------------------------------------------------------
// Edge MLP: out = LN( relu( relu( [n[src]|n[dst]|e] @ W1 + b1 ) @ W2 + b2 ) @ W3 + b3 )
// Strategy: bf16 WMMA (v_wmma_f32_16x16x32_bf16) with f32 accum; weights
// pre-transposed+converted to bf16 [N][K] in d_ws; activations staged in LDS.
// ---------------------------------------------------------------------------

typedef __attribute__((ext_vector_type(16))) __bf16       v16bf;
typedef __attribute__((ext_vector_type(8)))  float        v8f;
typedef __attribute__((ext_vector_type(4)))  unsigned int v4u;

#define NODE_DIM 128
#define EDGE_DIM 128
#define IN_DIM   384
#define HIDDEN   256
#define OUT_DIM  128
#define M_TILE   128
#define THREADS  256

#define LDX 392   // 384 + 8 pad (bf16 elems) -> 196 dwords/row, stride%64 = 4: conflict-free
#define LDH 264   // 256 + 8 pad (bf16 elems)
#define LDO 132   // 128 + 4 pad (f32  elems)

// ws layout (bf16 element offsets)
#define WT1_OFF 0         // [256][384]
#define WT2_OFF 98304     // [256][256]
#define WT3_OFF 163840    // [128][256]

static __device__ __forceinline__ unsigned short f32_to_bf16(float f) {
    unsigned int u = __float_as_uint(f);
    u += 0x7FFFu + ((u >> 16) & 1u);   // round-to-nearest-even
    return (unsigned short)(u >> 16);
}

// ---------------------------------------------------------------------------
// Pre-pass: convert fp32 weights to bf16, transposed to [N][K] so a WMMA
// B-fragment (two 8-K runs per lane for fixed column) is two 16B loads.
// ---------------------------------------------------------------------------
__global__ void convert_weights_kernel(const float* __restrict__ W1,
                                       const float* __restrict__ W2,
                                       const float* __restrict__ W3,
                                       unsigned short* __restrict__ ws) {
    int i = blockIdx.x * blockDim.x + threadIdx.x;
    if (i < 256 * 384) {                       // W1: [384][256] -> Wt1[n][k]
        int n = i / 384, k = i % 384;
        ws[WT1_OFF + i] = f32_to_bf16(W1[k * 256 + n]);
        return;
    }
    i -= 256 * 384;
    if (i < 256 * 256) {                       // W2: [256][256] -> Wt2[n][k]
        int n = i / 256, k = i % 256;
        ws[WT2_OFF + n * 256 + k] = f32_to_bf16(W2[k * 256 + n]);
        return;
    }
    i -= 256 * 256;
    if (i < 128 * 256) {                       // W3: [256][128] -> Wt3[n][k]
        int n = i / 256, k = i % 256;
        ws[WT3_OFF + n * 256 + k] = f32_to_bf16(W3[k * 128 + n]);
    }
}

// ---------------------------------------------------------------------------
// One GEMM layer for one wave's 16-row tile, sweeping `ntiles` column tiles.
// A (16x32 bf16) lane layout: row = lane&15; K chunks 8*(lane>>4)+[0..8) and +16.
// B (32x16 bf16) lane layout: col = lane&15; K chunks 8*(lane>>4)+[0..8) and +16.
// C (16x16 f32): VGPR r -> row r + 8*(lane>>4); col = lane&15.
// ---------------------------------------------------------------------------
static __device__ __forceinline__ void gemm_layer(
    const unsigned short* __restrict__ Xs, int ldx, int K,
    const unsigned short* __restrict__ Wt,          // [N][K] bf16 (global/L2)
    const float* __restrict__ bias,
    int ntiles, int rowtile, int lane, bool relu,
    unsigned short* __restrict__ Hbf, int ldh,      // bf16 output (layers 1,2)
    float* __restrict__ Hf, int ldo)                // f32 output  (layer 3)
{
    const int nl    = lane & 15;
    const int kb_a  = (lane >> 4) * 8;
    const int kb_b  = (lane >> 4) * 8;
    const int arowm = rowtile * 16 + (lane & 15);
    const int rbase = rowtile * 16 + ((lane >> 4) << 3);
    const unsigned short* arow = Xs + arowm * ldx + kb_a;

    for (int nt = 0; nt < ntiles; ++nt) {
        const int n = nt * 16 + nl;
        const float bv = bias[n];
        v8f acc = {bv, bv, bv, bv, bv, bv, bv, bv};
        const unsigned short* wcol = Wt + n * K + kb_b;

        for (int k = 0; k < K; k += 32) {
            union { v16bf v; v4u u[2]; } A, B;
            A.u[0] = *(const v4u*)(arow + k);        // K = k+kb_a   .. +8
            A.u[1] = *(const v4u*)(arow + k + 16);   // K = k+kb_a+16.. +8
            B.u[0] = *(const v4u*)(wcol + k);        // K = k+kb_b   .. +8
            B.u[1] = *(const v4u*)(wcol + k + 16);   // K = k+kb_b+16.. +8
            acc = __builtin_amdgcn_wmma_f32_16x16x32_bf16(
                false, A.v, false, B.v, (short)0, acc, false, false);
        }
        #pragma unroll
        for (int r = 0; r < 8; ++r) {
            float v = acc[r];
            if (relu) v = fmaxf(v, 0.0f);
            const int row = rbase + r;
            if (Hbf) Hbf[row * ldh + nt * 16 + nl] = f32_to_bf16(v);
            else     Hf [row * ldo + nt * 16 + nl] = v;
        }
    }
}

// ---------------------------------------------------------------------------
__global__ void __launch_bounds__(THREADS)
edge_mlp_kernel(const float* __restrict__ nodef,
                const float* __restrict__ edgef,
                const long long* __restrict__ eidx,   // [2][nE] int64
                const float* __restrict__ b1, const float* __restrict__ b2,
                const float* __restrict__ b3,
                const float* __restrict__ gamma, const float* __restrict__ beta,
                const unsigned short* __restrict__ wt,
                float* __restrict__ out, int nE)
{
    __shared__ __align__(16) unsigned char s_bufA[M_TILE * LDX * 2]; // X0 / H2
    __shared__ __align__(16) unsigned char s_bufB[M_TILE * LDH * 2]; // H1 / H3(f32)
    __shared__ float s_mu[M_TILE];
    __shared__ float s_rstd[M_TILE];

    unsigned short* Xs = (unsigned short*)s_bufA;
    unsigned short* H1 = (unsigned short*)s_bufB;
    unsigned short* H2 = (unsigned short*)s_bufA;
    float*          H3 = (float*)s_bufB;            // 128*132*4 == sizeof(s_bufB)

    const int tid  = threadIdx.x;
    const int wave = tid >> 5;
    const int lane = tid & 31;
    const long long edge0 = (long long)blockIdx.x * M_TILE;

    // ---- Gather: 2 threads per edge, each covers 192 of the 384 input cols.
    {
        const int e     = tid >> 1;
        const int chalf = (tid & 1) * 192;
        const long long ge = edge0 + e;
        const bool valid = ge < (long long)nE;
        long long s = 0, d = 0;
        if (valid) { s = eidx[ge]; d = eidx[(long long)nE + ge]; }
        const float* p0 = nodef + s * NODE_DIM;
        const float* p1 = nodef + d * NODE_DIM;
        const float* p2 = edgef + ge * EDGE_DIM;
        unsigned short* xr = Xs + e * LDX + chalf;
        for (int c = 0; c < 192; c += 2) {
            const int cc = chalf + c;
            float v0, v1;
            if (!valid)        { v0 = 0.0f;         v1 = 0.0f; }
            else if (cc < 128) { v0 = p0[cc];       v1 = p0[cc + 1]; }
            else if (cc < 256) { v0 = p1[cc - 128]; v1 = p1[cc - 127]; }
            else               { v0 = p2[cc - 256]; v1 = p2[cc - 255]; }
            unsigned int pk = (unsigned int)f32_to_bf16(v0)
                            | ((unsigned int)f32_to_bf16(v1) << 16);
            *(unsigned int*)(xr + c) = pk;
        }
    }
    __syncthreads();

    // ---- Layer 1: [128x384] @ [384x256] + b1, relu -> H1 (bf16)
    gemm_layer(Xs, LDX, IN_DIM, wt + WT1_OFF, b1, 16, wave, lane, true,
               H1, LDH, nullptr, 0);
    __syncthreads();

    // ---- Layer 2: [128x256] @ [256x256] + b2, relu -> H2 (bf16, reuses bufA)
    gemm_layer(H1, LDH, HIDDEN, wt + WT2_OFF, b2, 16, wave, lane, true,
               H2, LDH, nullptr, 0);
    __syncthreads();

    // ---- Layer 3: [128x256] @ [256x128] + b3 -> H3 (f32, reuses bufB)
    gemm_layer(H2, LDH, HIDDEN, wt + WT3_OFF, b3, 8, wave, lane, false,
               nullptr, 0, H3, LDO);
    __syncthreads();

    // ---- LayerNorm stats: one thread per row.
    if (tid < M_TILE) {
        const float* hr = H3 + tid * LDO;
        float sum = 0.0f;
        for (int c = 0; c < OUT_DIM; ++c) sum += hr[c];
        const float mu = sum * (1.0f / OUT_DIM);
        float vs = 0.0f;
        for (int c = 0; c < OUT_DIM; ++c) { float dd = hr[c] - mu; vs += dd * dd; }
        s_mu[tid]   = mu;
        s_rstd[tid] = rsqrtf(vs * (1.0f / OUT_DIM) + 1e-5f);
    }
    __syncthreads();

    // ---- Normalize + affine, coalesced store.
    for (int idx = tid; idx < M_TILE * OUT_DIM; idx += THREADS) {
        const int row = idx >> 7;
        const int col = idx & 127;
        const long long ge = edge0 + row;
        if (ge < (long long)nE) {
            float v = (H3[row * LDO + col] - s_mu[row]) * s_rstd[row]
                      * gamma[col] + beta[col];
            out[ge * OUT_DIM + col] = v;
        }
    }
}

// ---------------------------------------------------------------------------
extern "C" void kernel_launch(void* const* d_in, const int* in_sizes, int n_in,
                              void* d_out, int out_size, void* d_ws, size_t ws_size,
                              hipStream_t stream) {
    const float*     nodef = (const float*)d_in[0];
    const float*     edgef = (const float*)d_in[1];
    const long long* eidx  = (const long long*)d_in[2];
    const float*     W1    = (const float*)d_in[3];
    const float*     b1    = (const float*)d_in[4];
    const float*     W2    = (const float*)d_in[5];
    const float*     b2    = (const float*)d_in[6];
    const float*     W3    = (const float*)d_in[7];
    const float*     b3    = (const float*)d_in[8];
    const float*     gamma = (const float*)d_in[9];
    const float*     beta  = (const float*)d_in[10];

    const int nE = in_sizes[1] / EDGE_DIM;             // 640000
    unsigned short* wt = (unsigned short*)d_ws;        // needs ~393 KB

    const int wtot = 256 * 384 + 256 * 256 + 128 * 256;
    convert_weights_kernel<<<(wtot + 255) / 256, 256, 0, stream>>>(W1, W2, W3, wt);

    const int nblocks = (nE + M_TILE - 1) / M_TILE;    // 5000
    edge_mlp_kernel<<<nblocks, THREADS, 0, stream>>>(
        nodef, edgef, eidx, b1, b2, b3, gamma, beta, wt, (float*)d_out, nE);
}
